// ToeplitzlikeLinear_33036888441442
// MI455X (gfx1250) — compile-verified, matrix operand order checked by reading, not run on hardware
//
#include <hip/hip_runtime.h>
#include <stdint.h>

// Toeplitz-like linear layer for MI455X (gfx1250, wave32, WMMA).
// out = U @ M^T (+bias), M_s = sum_r L_G L_H^T built by diagonal prefix-scan.
// GEMM: bf16 WMMA with double-buffered async global->LDS tile staging.

typedef __attribute__((ext_vector_type(16))) __bf16 bf16x16;
typedef __attribute__((ext_vector_type(8)))  __bf16 bf16x8;
typedef __attribute__((ext_vector_type(8)))  float  f32x8;

#define IN_N   1024
#define OUT_N  4096
#define NSTACK 4
#define RANK   4
#define NTOK   4096            // 2 * 2048 tokens

#define KSTEP   32
#define NKSTEP  (IN_N / KSTEP) // 32
#define ROW_B   80             // LDS row stride: 32 bf16 = 64B data + 16B pad
#define A_TILE_B (128 * ROW_B) // 10240 B
#define B_TILE_B (256 * ROW_B) // 20480 B
#define BUF_B    (A_TILE_B + B_TILE_B) // 30720 B per pipeline stage

// ---------------------------------------------------------------------------
// Kernel 1: build M in bf16, layout Mbf[s][t][m] (m contiguous).
// M[t,m] = sum_r sum_{j<=min(t,m)} G[t-j] H[m-j]  (diagonal prefix scan).
// ---------------------------------------------------------------------------
__global__ __launch_bounds__(256) void build_m_kernel(
    const float* __restrict__ G, const float* __restrict__ H,
    __bf16* __restrict__ Mbf)
{
  const int lane = threadIdx.x & 31;
  const int wid  = threadIdx.x >> 5;
  const int gw   = blockIdx.x * 8 + wid;          // one wave per diagonal
  const int s      = gw >> 11;
  const int rest   = gw & 2047;
  const int region = rest >> 10;                  // 0: t>=m, 1: m>t
  const int delta  = (rest & 1023) + region;
  if (delta > 1023) return;

  const float* __restrict__ Pa = (region ? H : G) + s * (RANK * IN_N); // [delta+i]
  const float* __restrict__ Pb = (region ? G : H) + s * (RANK * IN_N); // [i]

  const int len = IN_N - delta;
  const int i0  = lane * 32;
  const int i1  = (i0 + 32 < len) ? (i0 + 32) : len;

  float partial = 0.f;
  for (int i = i0; i < i1; ++i) {
    float q = 0.f;
#pragma unroll
    for (int r = 0; r < RANK; ++r)
      q = fmaf(Pa[r * IN_N + delta + i], Pb[r * IN_N + i], q);
    partial += q;
  }

  float v = partial;                              // inclusive wave32 scan
#pragma unroll
  for (int off = 1; off < 32; off <<= 1) {
    float nb = __shfl_up(v, off, 32);
    if (lane >= off) v += nb;
  }
  float run = v - partial;                        // exclusive prefix

  for (int i = i0; i < i1; ++i) {
    float q = 0.f;
#pragma unroll
    for (int r = 0; r < RANK; ++r)
      q = fmaf(Pa[r * IN_N + delta + i], Pb[r * IN_N + i], q);
    run += q;
    const int t = region ? i : (delta + i);
    const int m = region ? (delta + i) : i;
    Mbf[(((size_t)s * IN_N + t) << 10) + m] = (__bf16)run;
  }
}

// ---------------------------------------------------------------------------
// Kernel 2: convert U (4096 x 1024 f32) -> bf16, 8 elems/thread.
// ---------------------------------------------------------------------------
__global__ __launch_bounds__(256) void cvt_kernel(
    const float* __restrict__ x, __bf16* __restrict__ Ubf)
{
  const size_t gid = (size_t)blockIdx.x * blockDim.x + threadIdx.x;
  const float4 f0 = ((const float4*)x)[gid * 2 + 0];
  const float4 f1 = ((const float4*)x)[gid * 2 + 1];
  bf16x8 o;
  o[0] = (__bf16)f0.x; o[1] = (__bf16)f0.y; o[2] = (__bf16)f0.z; o[3] = (__bf16)f0.w;
  o[4] = (__bf16)f1.x; o[5] = (__bf16)f1.y; o[6] = (__bf16)f1.z; o[7] = (__bf16)f1.w;
  ((bf16x8*)Ubf)[gid] = o;
}

// ---------------------------------------------------------------------------
// Kernel 3: GEMM out[b,n] = sum_k Ubf[b,k]*Mbf[n][k] + bias[n]
// Block tile 128x256 (8 waves, wave tile 64x64 = 4x4 WMMA), K-step 32.
// Tiles staged to LDS with GLOBAL_LOAD_ASYNC_TO_LDS_B128 (ASYNCcnt),
// double-buffered, one barrier per K-step.
// ---------------------------------------------------------------------------
__device__ __forceinline__ void async_cp16(uint32_t lds_off, uint64_t gaddr) {
  asm volatile("global_load_async_to_lds_b128 %0, %1, off"
               :: "v"(lds_off), "v"(gaddr) : "memory");
}

__device__ __forceinline__ bf16x16 lds_frag(const unsigned char* p) {
  bf16x8 lo = *(const bf16x8*)(p);       // K = base .. base+7
  bf16x8 hh = *(const bf16x8*)(p + 32);  // K = base+16 .. base+23
  bf16x16 r;
#pragma unroll
  for (int e = 0; e < 8; ++e) { r[e] = lo[e]; r[e + 8] = hh[e]; }
  return r;
}

__global__ __launch_bounds__(256) void gemm_kernel(
    const __bf16* __restrict__ U, const __bf16* __restrict__ M,
    const float* __restrict__ bias, float* __restrict__ out)
{
  __shared__ __align__(16) unsigned char smem[2 * BUF_B];

  const int tid  = threadIdx.x;
  const int lane = tid & 31;
  const int w    = tid >> 5;
  const int wm   = w & 1;          // 2 waves along M
  const int wn   = w >> 1;         // 4 waves along N
  const int hi   = lane >> 4;      // K-half selector
  const int lr   = lane & 15;      // row (A) / col (B) within 16x16 tile

  const int rowBlk = blockIdx.y * 128;
  const int colBlk = blockIdx.x * 256;

  // Flat->LDS: low 32 bits of the flat address are the wave-relative offset.
  const uint32_t smem0 = (uint32_t)(uintptr_t)&smem[0];

  // ---- async-copy assignment: 6 x 16B per thread per K-step ----
  const int crow = tid >> 2;       // 0..63
  const int cseg = tid & 3;        // which 16B of the 64B row chunk
  const uint64_t gA = (uint64_t)(uintptr_t)U +
                      ((uint64_t)(rowBlk + crow) << 11) + (uint64_t)cseg * 16;
  const uint64_t gB = (uint64_t)(uintptr_t)M +
                      ((uint64_t)(colBlk + crow) << 11) + (uint64_t)cseg * 16;
  const uint32_t lA = (uint32_t)(crow * ROW_B + cseg * 16);
  const uint32_t lB = (uint32_t)(A_TILE_B + crow * ROW_B + cseg * 16);

  // ---- fragment LDS offsets ----
  uint32_t aOff[4], bOff[4];
#pragma unroll
  for (int i = 0; i < 4; ++i)
    aOff[i] = (uint32_t)((wm * 64 + i * 16 + lr) * ROW_B + hi * 16);
#pragma unroll
  for (int j = 0; j < 4; ++j)
    bOff[j] = (uint32_t)(A_TILE_B + (wn * 64 + j * 16 + lr) * ROW_B + hi * 16);

  f32x8 acc[4][4] = {};

  // ---- pipeline prologue: stage K-step 0 into buffer 0 ----
  {
    const uint32_t lbase = smem0;
    async_cp16(lbase + lA,                gA);
    async_cp16(lbase + lA + 64 * ROW_B,   gA + (64ull << 11));
#pragma unroll
    for (int it = 0; it < 4; ++it)
      async_cp16(lbase + lB + it * 64 * ROW_B, gB + ((uint64_t)(it * 64) << 11));
  }

#pragma unroll 1
  for (int ks = 0; ks < NKSTEP; ++ks) {
    const int cur = ks & 1;

    // my copies for step ks are done; barrier publishes everyone's tile and
    // guarantees the other buffer's previous consumers have finished.
    asm volatile("s_wait_asynccnt 0x0" ::: "memory");
    __syncthreads();

    if (ks + 1 < NKSTEP) {                       // stage next K-step
      const uint32_t lbase = smem0 + (uint32_t)((1 - cur) * BUF_B);
      const uint64_t koff  = (uint64_t)(ks + 1) * KSTEP * 2;
      async_cp16(lbase + lA,              gA + koff);
      async_cp16(lbase + lA + 64 * ROW_B, gA + koff + (64ull << 11));
#pragma unroll
      for (int it = 0; it < 4; ++it)
        async_cp16(lbase + lB + it * 64 * ROW_B,
                   gB + koff + ((uint64_t)(it * 64) << 11));
    }

    const unsigned char* bufp = smem + cur * BUF_B;
    bf16x16 fa[4], fb[4];
#pragma unroll
    for (int i = 0; i < 4; ++i) fa[i] = lds_frag(bufp + aOff[i]);
#pragma unroll
    for (int j = 0; j < 4; ++j) fb[j] = lds_frag(bufp + bOff[j]);
#pragma unroll
    for (int i = 0; i < 4; ++i)
#pragma unroll
      for (int j = 0; j < 4; ++j)
        acc[i][j] = __builtin_amdgcn_wmma_f32_16x16x32_bf16(
            false, fa[i], false, fb[j], (short)0, acc[i][j], false, false);
  }

  // Epilogue: lane holds col n = lr, rows hi*8 + r; fuse bias add.
#pragma unroll
  for (int j = 0; j < 4; ++j) {
    const int ng = colBlk + wn * 64 + j * 16 + lr;
    const float bv = bias[ng];
#pragma unroll
    for (int i = 0; i < 4; ++i) {
      const int rb = rowBlk + wm * 64 + i * 16 + hi * 8;
#pragma unroll
      for (int r = 0; r < 8; ++r)
        out[(size_t)(rb + r) * OUT_N + ng] = acc[i][j][r] + bv;
    }
  }
}

// ---------------------------------------------------------------------------
extern "C" void kernel_launch(void* const* d_in, const int* in_sizes, int n_in,
                              void* d_out, int out_size, void* d_ws, size_t ws_size,
                              hipStream_t stream) {
  (void)in_sizes; (void)n_in; (void)out_size; (void)ws_size;
  const float* x    = (const float*)d_in[0];
  const float* G    = (const float*)d_in[1];
  const float* H    = (const float*)d_in[2];
  const float* bias = (const float*)d_in[3];
  float* out = (float*)d_out;

  // Workspace partition: M bf16 (8 MB) | U bf16 (8 MB)
  __bf16* Mbf = (__bf16*)d_ws;
  __bf16* Ubf = (__bf16*)((char*)d_ws +
                          (size_t)NSTACK * IN_N * IN_N * sizeof(__bf16));

  build_m_kernel<<<1024, 256, 0, stream>>>(G, H, Mbf);
  cvt_kernel<<<(NTOK * IN_N / 8) / 256, 256, 0, stream>>>(x, Ubf);
  gemm_kernel<<<dim3(OUT_N / 256, NTOK / 128), 256, 0, stream>>>(Ubf, Mbf, bias, out);
}
